// Group_SA_Linear_80882824119002
// MI455X (gfx1250) — matrix operand order repaired and checked by reading, not compiled, not run on hardware
//
#include <hip/hip_runtime.h>
#include <hip/hip_bf16.h>
#include <math.h>

// ---------------------------------------------------------------------------
// CDNA5 (gfx1250, wave32) implementation of the grouped linear-attention block.
// Heavy GEMMs run on v_wmma_f32_16x16x32_bf16 (fp32->bf16 staged through LDS,
// f32 accumulation), with double-buffered LDS tiles and vectorized staging.
// ---------------------------------------------------------------------------

typedef __bf16 bf16_t;
typedef __attribute__((ext_vector_type(16))) __bf16 v16bf;
typedef __attribute__((ext_vector_type(8)))  __bf16 v8bf;
typedef __attribute__((ext_vector_type(4)))  __bf16 v4bf;
typedef __attribute__((ext_vector_type(8)))  float  v8f;

#define GPv   4
#define Bsz   8
#define Cdim  512
#define Ndim  4096
#define NG    1024      // N / GP
#define Gtot  32        // GP * B
#define Fdim  2048

// GEMM tiling: 8 waves (256 thr), 128x128 block tile, 32x64 wave tile, K-step 32.
#define TM 128
#define TN 128
#define TK 32
#define LDSS 40         // LDS row stride in bf16 elements (80B: 16B- and 8B-aligned rows)

enum { EPI_NONE = 0, EPI_PHI = 1, EPI_BIAS_RELU = 2, EPI_BIAS_ADD_RES = 3, EPI_ZPHI = 4 };

__device__ __forceinline__ v8f wmma_bf16(v16bf a, v16bf b, v8f c) {
    return __builtin_amdgcn_wmma_f32_16x16x32_bf16(false, a, false, b, (short)0, c, false, false);
}

// Batched GEMM:  Y[bz][m][col] = epi( scale * sum_k A[bz%bmodA][m][k] * B[bz][col][k] )
// A is (M,K) row-major.  BT=0: Bsrc is (Ncol,K) row-major.  BT=1: Bsrc is (K,Ncol).
template <int EPI, int BT>
__global__ __launch_bounds__(256) void gemm_wmma_kernel(
    const float* __restrict__ A, int lda, long long sA, int bmodA,
    const float* __restrict__ Bsrc, int ldb, long long sB,
    float* __restrict__ Y, int ldy, long long sY,
    int K, float scale,
    const float* __restrict__ bias,
    const float* __restrict__ zvec, int zld,
    const float* __restrict__ mmat, long long sM, int ldm,
    const float* __restrict__ res, long long sR, int ldr)
{
    __shared__ __align__(16) bf16_t As[2][TM * LDSS];
    __shared__ __align__(16) bf16_t Bs[2][TN * LDSS];

    const int tid   = threadIdx.x;
    const int wave  = tid >> 5;
    const int lane  = tid & 31;
    const int hi    = lane >> 4;
    const int l16   = lane & 15;
    const int mrow0 = 32 * (wave >> 1);   // wave tile: 32 rows x 64 cols
    const int ncol0 = 64 * (wave & 1);

    const int bz      = blockIdx.z;
    const int colbase = blockIdx.x * TN;
    const int mbase   = blockIdx.y * TM;

    const float* Ab = A + (long long)(bz % bmodA) * sA;
    const float* Bb = Bsrc + (long long)bz * sB;

    // ---- per-thread staging assignments (float4 granularity) ----
    const float* pa[4]; int sa_off[4];
    const float* pb[4]; int sb_off[4];
    #pragma unroll
    for (int i = 0; i < 4; ++i) {
        int q = tid + i * 256;
        int m = q >> 3, k4 = q & 7;          // 8 float4 per 32-wide row
        pa[i]     = Ab + (long long)(mbase + m) * lda + k4 * 4;
        sa_off[i] = m * LDSS + k4 * 4;
        if (BT) {                            // source (K, Ncol): 32 float4 per 128-wide row
            int k = q >> 5, c4 = q & 31;
            pb[i]     = Bb + (long long)k * ldb + colbase + c4 * 4;
            sb_off[i] = (c4 * 4) * LDSS + k; // transposed scatter, +j*LDSS
        } else {                             // source (Ncol, K)
            int n = q >> 3, kb = q & 7;
            pb[i]     = Bb + (long long)(colbase + n) * ldb + kb * 4;
            sb_off[i] = n * LDSS + kb * 4;
        }
    }

    float4 ra[4], rb[4];
    #pragma unroll
    for (int i = 0; i < 4; ++i) { ra[i] = *(const float4*)pa[i]; rb[i] = *(const float4*)pb[i]; }

    auto store_tile = [&](int buf) {
        #pragma unroll
        for (int i = 0; i < 4; ++i) {
            v4bf a4;
            a4[0] = (bf16_t)ra[i].x; a4[1] = (bf16_t)ra[i].y;
            a4[2] = (bf16_t)ra[i].z; a4[3] = (bf16_t)ra[i].w;
            *(v4bf*)&As[buf][sa_off[i]] = a4;               // ds_store_b64
            if (BT) {
                Bs[buf][sb_off[i] + 0 * LDSS] = (bf16_t)rb[i].x;
                Bs[buf][sb_off[i] + 1 * LDSS] = (bf16_t)rb[i].y;
                Bs[buf][sb_off[i] + 2 * LDSS] = (bf16_t)rb[i].z;
                Bs[buf][sb_off[i] + 3 * LDSS] = (bf16_t)rb[i].w;
            } else {
                v4bf b4;
                b4[0] = (bf16_t)rb[i].x; b4[1] = (bf16_t)rb[i].y;
                b4[2] = (bf16_t)rb[i].z; b4[3] = (bf16_t)rb[i].w;
                *(v4bf*)&Bs[buf][sb_off[i]] = b4;
            }
        }
    };
    store_tile(0);

    v8f acc[2][4] = {};
    const int nk = K / TK;
    for (int kt = 0; kt < nk; ++kt) {
        __syncthreads();
        const int cur = kt & 1;
        if (kt + 1 < nk) {                   // issue next tile's loads early
            #pragma unroll
            for (int i = 0; i < 4; ++i) {
                pa[i] += TK;
                pb[i] += BT ? (long long)TK * ldb : (long long)TK;
                ra[i] = *(const float4*)pa[i];
                rb[i] = *(const float4*)pb[i];
            }
        }
        // ---- fragments + 8 WMMAs ----
        // A frag (16x32): K(e) = (e&8)*2 + 8*hi + (e&7)
        v16bf af[2];
        #pragma unroll
        for (int ms = 0; ms < 2; ++ms) {
            const bf16_t* ap = &As[cur][(mrow0 + 16 * ms + l16) * LDSS + 8 * hi];
            v8bf lo = *(const v8bf*)ap;
            v8bf hh = *(const v8bf*)(ap + 16);
            #pragma unroll
            for (int e = 0; e < 8; ++e) { af[ms][e] = lo[e]; af[ms][8 + e] = hh[e]; }
        }
        #pragma unroll
        for (int nt = 0; nt < 4; ++nt) {     // B frag: K(e) = (e&8)*2 + 8*hi + (e&7)
            v16bf bfv;
            const bf16_t* bp = &Bs[cur][(ncol0 + 16 * nt + l16) * LDSS + 8 * hi];
            v8bf lo = *(const v8bf*)bp;
            v8bf hh = *(const v8bf*)(bp + 16);
            #pragma unroll
            for (int e = 0; e < 8; ++e) { bfv[e] = lo[e]; bfv[8 + e] = hh[e]; }
            acc[0][nt] = wmma_bf16(af[0], bfv, acc[0][nt]);
            acc[1][nt] = wmma_bf16(af[1], bfv, acc[1][nt]);
        }
        if (kt + 1 < nk) store_tile(cur ^ 1);
    }

    // ---- epilogue: D layout row = r + 8*hi, col = 16*nt + l16 ----
    float* Yb = Y + (long long)bz * sY;
    #pragma unroll
    for (int ms = 0; ms < 2; ++ms) {
        #pragma unroll
        for (int nt = 0; nt < 4; ++nt) {
            int col = colbase + ncol0 + 16 * nt + l16;
            #pragma unroll
            for (int r = 0; r < 8; ++r) {
                int m   = mbase + mrow0 + 16 * ms + r + 8 * hi;
                float v = acc[ms][nt][r] * scale;
                if constexpr (EPI == EPI_PHI)            v = fmaxf(v, 0.0f) + 1.0f;
                else if constexpr (EPI == EPI_BIAS_RELU) v = fmaxf(v + bias[col], 0.0f);
                else if constexpr (EPI == EPI_BIAS_ADD_RES)
                    v = v + bias[col] + res[(long long)bz * sR + (long long)m * ldr + col];
                else if constexpr (EPI == EPI_ZPHI)
                    v = v * zvec[(long long)bz * zld + col] *
                        mmat[(long long)bz * sM + (long long)m * ldm + col];
                Yb[(long long)m * ldy + col] = v;
            }
        }
    }
}

// ---------------- helper kernels (fp32, bandwidth-bound) ----------------

__global__ void regroup1_kernel(float* __restrict__ gx, const float* __restrict__ feat) {
    long long idx = (long long)blockIdx.x * blockDim.x + threadIdx.x;
    int i = idx % NG;
    int c = (int)((idx / NG) % Cdim);
    int G = (int)(idx / ((long long)NG * Cdim));
    int j = G / Bsz, b = G % Bsz;
    gx[idx] = feat[((long long)b * Cdim + c) * Ndim + (long long)i * GPv + j];
}

__global__ void regroup2_kernel(float* __restrict__ gx2, const float* __restrict__ f) {
    long long idx = (long long)blockIdx.x * blockDim.x + threadIdx.x;
    int i2 = idx % NG;
    int c  = (int)((idx / NG) % Cdim);
    int G2 = (int)(idx / ((long long)NG * Cdim));
    int j2 = G2 / Bsz, b = G2 % Bsz;
    int srcG = (i2 & 3) * Bsz + b;
    int srcn = j2 * (NG / GPv) + (i2 >> 2);
    gx2[idx] = f[((long long)srcG * Cdim + c) * NG + srcn];
}

__global__ void final_kernel(float* __restrict__ out, const float* __restrict__ outfirst,
                             const float* __restrict__ gx2) {
    long long idx = (long long)blockIdx.x * blockDim.x + threadIdx.x;
    int n = idx % Ndim;
    int c = (int)((idx / Ndim) % Cdim);
    int b = (int)(idx / ((long long)Ndim * Cdim));
    int j2 = n >> 10, i2 = n & (NG - 1);
    float v;
    if (j2 == 0) v = outfirst[((long long)b * Cdim + c) * NG + i2];
    else         v = gx2[((long long)(j2 * Bsz + b) * Cdim + c) * NG + i2];
    out[idx] = v;
}

// out[row] = scale * sum(in[row, 0:L])
__global__ __launch_bounds__(256) void row_reduce_kernel(float* __restrict__ out,
                                                         const float* __restrict__ in,
                                                         int L, float scale) {
    __shared__ float red[256];
    long long row = blockIdx.x;
    const float* p = in + row * L;
    float s = 0.f;
    for (int i = threadIdx.x; i < L; i += 256) s += p[i];
    red[threadIdx.x] = s; __syncthreads();
    for (int st = 128; st > 0; st >>= 1) {
        if (threadIdx.x < st) red[threadIdx.x] += red[threadIdx.x + st];
        __syncthreads();
    }
    if (threadIdx.x == 0) out[row] = red[0] * scale;
}

// alpha[g,n] = softmax_n( sum_c qg[g,c]*q[g,c,n] ) * n    (block = one group, 1024 thr)
__global__ __launch_bounds__(1024) void alpha_kernel(float* __restrict__ alpha,
                                                     const float* __restrict__ q,
                                                     const float* __restrict__ qg,
                                                     int C, int n) {
    __shared__ float red[1024];
    int g = blockIdx.x, t = threadIdx.x;
    const float* qp = q + (long long)g * C * n + t;
    const float* gp = qg + (long long)g * C;
    float s = 0.f;
    for (int c = 0; c < C; ++c) s += gp[c] * qp[(long long)c * n];
    red[t] = s; __syncthreads();
    for (int st = 512; st > 0; st >>= 1) { if (t < st) red[t] = fmaxf(red[t], red[t + st]); __syncthreads(); }
    float mx = red[0]; __syncthreads();
    float e = expf(s - mx);
    red[t] = e; __syncthreads();
    for (int st = 512; st > 0; st >>= 1) { if (t < st) red[t] += red[t + st]; __syncthreads(); }
    alpha[(long long)g * n + t] = e / red[0] * (float)n;
}

// z[g,n] = 1 / (sum_c q[g,c,n]*ksum[g%kmod,c] + 1e-6)
__global__ __launch_bounds__(1024) void z_kernel(float* __restrict__ z,
                                                 const float* __restrict__ q,
                                                 const float* __restrict__ ksum,
                                                 int C, int n, int kmod) {
    int g = blockIdx.x, t = threadIdx.x;
    const float* qp = q + (long long)g * C * n + t;
    const float* kp = ksum + (long long)(g % kmod) * C;
    float s = 0.f;
    for (int c = 0; c < C; ++c) s += kp[c] * qp[(long long)c * n];
    z[(long long)g * n + t] = 1.0f / (s + 1e-6f);
}

// out[g,c,n] = in[g,c,n] * alpha[g,n]   (in-place safe)
__global__ void colscale_kernel(float* __restrict__ out, const float* __restrict__ in,
                                const float* __restrict__ alpha, int C, int n) {
    long long idx = (long long)blockIdx.x * blockDim.x + threadIdx.x;
    long long per = (long long)C * n;
    int g = (int)(idx / per);
    int t = (int)(idx % n);
    out[idx] = in[idx] * alpha[(long long)g * n + t];
}

// dst[z][c][r] = a[z][r][c] (+ b[z][r][c]) ; a,b : (batch,R,Cc)
__global__ __launch_bounds__(256) void transpose_add_kernel(float* __restrict__ dst,
                                                            const float* __restrict__ a,
                                                            const float* __restrict__ b,
                                                            int R, int Cc) {
    __shared__ float tile[32][33];
    int zb = blockIdx.z;
    long long boff = (long long)zb * R * Cc;
    int c0 = blockIdx.x * 32, r0 = blockIdx.y * 32;
    int tx = threadIdx.x & 31, ty = threadIdx.x >> 5;
    for (int i = 0; i < 32; i += 8) {
        long long src = boff + (long long)(r0 + ty + i) * Cc + (c0 + tx);
        float v = a[src];
        if (b) v += b[src];
        tile[ty + i][tx] = v;
    }
    __syncthreads();
    for (int i = 0; i < 32; i += 8)
        dst[boff + (long long)(c0 + ty + i) * R + (r0 + tx)] = tile[tx][ty + i];
}

// LayerNorm over contiguous rows of length C==512 (256 thr, 2 elems/thr)
__global__ __launch_bounds__(256) void ln_rows_kernel(float* __restrict__ dst,
                                                      const float* __restrict__ src,
                                                      const float* __restrict__ gam,
                                                      const float* __restrict__ bet,
                                                      int C, int relu) {
    __shared__ float red[256];
    long long row = blockIdx.x;
    const float* p = src + row * C;
    float* o = dst + row * C;
    int t = threadIdx.x;
    float x0 = p[t], x1 = p[t + 256];
    red[t] = x0 + x1; __syncthreads();
    for (int st = 128; st > 0; st >>= 1) { if (t < st) red[t] += red[t + st]; __syncthreads(); }
    float mu = red[0] / (float)C; __syncthreads();
    float d0 = x0 - mu, d1 = x1 - mu;
    red[t] = d0 * d0 + d1 * d1; __syncthreads();
    for (int st = 128; st > 0; st >>= 1) { if (t < st) red[t] += red[t + st]; __syncthreads(); }
    float rstd = rsqrtf(red[0] / (float)C + 1e-6f);
    float y0 = d0 * rstd * gam[t] + bet[t];
    float y1 = d1 * rstd * gam[t + 256] + bet[t + 256];
    if (relu) { y0 = fmaxf(y0, 0.f); y1 = fmaxf(y1, 0.f); }
    o[t] = y0; o[t + 256] = y1;
}

// yfirst[b] = mean_g(y[g]) * phif[b]
__global__ void ymean_phifirst_kernel(float* __restrict__ yfirst, const float* __restrict__ y,
                                      const float* __restrict__ phif, int Grest, int Bn,
                                      long long per) {
    long long idx = (long long)blockIdx.x * blockDim.x + threadIdx.x;
    if (idx >= per) return;
    float s = 0.f;
    for (int g = 0; g < Grest; ++g) s += y[(long long)g * per + idx];
    s /= (float)Grest;
    for (int b = 0; b < Bn; ++b)
        yfirst[(long long)b * per + idx] = s * phif[(long long)b * per + idx];
}

// ---------------------------------------------------------------------------

extern "C" void kernel_launch(void* const* d_in, const int* in_sizes, int n_in,
                              void* d_out, int out_size, void* d_ws, size_t ws_size,
                              hipStream_t stream) {
    const float* feat  = (const float*)d_in[0];
    const float* tw_qk = (const float*)d_in[1];
    const float* tw_v  = (const float*)d_in[2];
    const float* tw_phi= (const float*)d_in[3];
    const float* tg1   = (const float*)d_in[4];
    const float* tb1   = (const float*)d_in[5];
    const float* tf1w  = (const float*)d_in[6];
    const float* tf1b  = (const float*)d_in[7];
    const float* tf2w  = (const float*)d_in[8];
    const float* tf2b  = (const float*)d_in[9];
    const float* tg2   = (const float*)d_in[10];
    const float* tb2   = (const float*)d_in[11];
    const float* cw_q  = (const float*)d_in[12];
    const float* cw_k  = (const float*)d_in[13];
    const float* cw_v  = (const float*)d_in[14];
    const float* cw_phi= (const float*)d_in[15];
    const float* cg1   = (const float*)d_in[16];
    const float* cb1   = (const float*)d_in[17];
    const float* cf1w  = (const float*)d_in[18];
    const float* cf1b  = (const float*)d_in[19];
    const float* cf2w  = (const float*)d_in[20];
    const float* cf2b  = (const float*)d_in[21];
    const float* cg2   = (const float*)d_in[22];
    const float* cb2   = (const float*)d_in[23];
    float* out = (float*)d_out;

    const long long PERf = (long long)Gtot * Cdim * NG;   // 16,777,216 floats
    const long long PERg = (long long)Cdim * NG;          // 524,288
    const long long KVg  = (long long)Cdim * Cdim;        // 262,144
    const long long KVf  = (long long)Gtot * KVg;         // 8,388,608
    const long long H1g  = (long long)NG * Fdim;          // 2,097,152
    const long long HLg  = (long long)NG * Cdim;          // 524,288

    float* W  = (float*)d_ws;
    float* S0 = W;               // gx / gx2
    float* S1 = W + 1 * PERf;    // q / hln / qr / hc_in / hsum_c
    float* S2 = W + 2 * PERf;    // v / y / f / phir / H1c / hco
    float* S3 = W + 3 * PERf;    // phix / H1 / k0(k0a) / yfirst / hcln
    float* S4 = W + 4 * PERf;    // qa / hin / hout / v0 / ycr
    float* S5 = W + 5 * PERf;    // hsum / phif / outfirst
    float* SKV = W + 6 * PERf;   // kvT / kvT0
    float* SM  = SKV + KVf;      // smalls
    float* qg     = SM;
    float* alpha  = SM + 16384;
    float* ksum   = SM + 49152;
    float* zb     = SM + 65536;
    float* k0mean = SM + 98304;
    float* alpha0 = SM + 102400;
    float* ksum0  = SM + 110592;
    float* z2     = SM + 114688;

    const float rsC = 1.0f / sqrtf((float)Cdim);
    const int BMOD_INF = 0x40000000;
    const float* NF = nullptr;

    // ================= SA block =================
    regroup1_kernel<<<PERf / 256, 256, 0, stream>>>(S0, feat);

    // q = phi(tw_qk @ gx), v = tw_v @ gx, phix = tw_phi @ gx
    dim3 gproj(NG / TN, Cdim / TM, Gtot);
    gemm_wmma_kernel<EPI_PHI, 1><<<gproj, 256, 0, stream>>>(
        tw_qk, Cdim, 0, 1, S0, NG, PERg, S1, NG, PERg, Cdim, 1.0f, NF, NF, 0, NF, 0, 0, NF, 0, 0);
    gemm_wmma_kernel<EPI_NONE, 1><<<gproj, 256, 0, stream>>>(
        tw_v, Cdim, 0, 1, S0, NG, PERg, S2, NG, PERg, Cdim, 1.0f, NF, NF, 0, NF, 0, 0, NF, 0, 0);
    gemm_wmma_kernel<EPI_NONE, 1><<<gproj, 256, 0, stream>>>(
        tw_phi, Cdim, 0, 1, S0, NG, PERg, S3, NG, PERg, Cdim, 1.0f, NF, NF, 0, NF, 0, 0, NF, 0, 0);

    row_reduce_kernel<<<Gtot * Cdim, 256, 0, stream>>>(qg, S1, NG, 1.0f / (float)NG);
    alpha_kernel<<<Gtot, 1024, 0, stream>>>(alpha, S1, qg, Cdim, NG);
    colscale_kernel<<<PERf / 256, 256, 0, stream>>>(S4, S1, alpha, Cdim, NG);   // qa
    row_reduce_kernel<<<Gtot * Cdim, 256, 0, stream>>>(ksum, S4, NG, 1.0f);

    // kvT[d,c] = sum_n v[d,n]*qa[c,n] / sqrt(C)
    dim3 gkv(Cdim / TN, Cdim / TM, Gtot);
    gemm_wmma_kernel<EPI_NONE, 0><<<gkv, 256, 0, stream>>>(
        S2, NG, PERg, BMOD_INF, S4, NG, PERg, SKV, Cdim, KVg, NG, rsC, NF, NF, 0, NF, 0, 0, NF, 0, 0);

    z_kernel<<<Gtot, 1024, 0, stream>>>(zb, S1, ksum, Cdim, NG, Gtot);

    // y[d,n] = (sum_c kvT[d,c]*q[c,n]) * z[n] * phix[d,n]   -> S2 (overwrites v)
    gemm_wmma_kernel<EPI_ZPHI, 1><<<gproj, 256, 0, stream>>>(
        SKV, Cdim, KVg, BMOD_INF, S1, NG, PERg, S2, NG, PERg, Cdim, 1.0f,
        NF, zb, NG, S3, PERg, NG, NF, 0, 0);

    // h = LN((gx + y)^T)
    dim3 gtr1(NG / 32, Cdim / 32, Gtot);
    transpose_add_kernel<<<gtr1, 256, 0, stream>>>(S4, S0, S2, Cdim, NG);       // hin (g,n,c)
    ln_rows_kernel<<<Gtot * NG, 256, 0, stream>>>(S1, S4, tg1, tb1, Cdim, 0);   // hln (g,n,c)

    // FFN in 4 chunks of 8 groups (H1 chunk reuses S3)
    for (int ch = 0; ch < 4; ++ch) {
        const float* hln = S1 + (long long)ch * 8 * HLg;
        float* hsum = S5 + (long long)ch * 8 * HLg;
        dim3 gf1(Fdim / TN, NG / TM, 8);
        gemm_wmma_kernel<EPI_BIAS_RELU, 0><<<gf1, 256, 0, stream>>>(
            hln, Cdim, HLg, BMOD_INF, tf1w, Cdim, 0, S3, Fdim, H1g, Cdim, 1.0f,
            tf1b, NF, 0, NF, 0, 0, NF, 0, 0);
        dim3 gf2(Cdim / TN, NG / TM, 8);
        gemm_wmma_kernel<EPI_BIAS_ADD_RES, 0><<<gf2, 256, 0, stream>>>(
            S3, Fdim, H1g, BMOD_INF, tf2w, Fdim, 0, hsum, Cdim, HLg, Fdim, 1.0f,
            tf2b, NF, 0, NF, 0, 0, hln, HLg, Cdim);
    }
    ln_rows_kernel<<<Gtot * NG, 256, 0, stream>>>(S4, S5, tg2, tb2, Cdim, 1);   // hout (g,n,c)
    dim3 gtr2(Cdim / 32, NG / 32, Gtot);
    transpose_add_kernel<<<gtr2, 256, 0, stream>>>(S2, S4, nullptr, NG, Cdim);  // f (g,c,n)

    // ================= cross block =================
    regroup2_kernel<<<PERf / 256, 256, 0, stream>>>(S0, S2);                    // gx2

    const float* first = S0;
    const float* rest  = S0 + 8 * PERg;
    dim3 gp24(NG / TN, Cdim / TM, 24);
    dim3 gp8(NG / TN, Cdim / TM, 8);
    gemm_wmma_kernel<EPI_PHI, 1><<<gp24, 256, 0, stream>>>(                     // qr
        cw_q, Cdim, 0, 1, rest, NG, PERg, S1, NG, PERg, Cdim, 1.0f, NF, NF, 0, NF, 0, 0, NF, 0, 0);
    gemm_wmma_kernel<EPI_PHI, 1><<<gp8, 256, 0, stream>>>(                      // k0
        cw_k, Cdim, 0, 1, first, NG, PERg, S3, NG, PERg, Cdim, 1.0f, NF, NF, 0, NF, 0, 0, NF, 0, 0);
    gemm_wmma_kernel<EPI_NONE, 1><<<gp8, 256, 0, stream>>>(                     // v0
        cw_v, Cdim, 0, 1, first, NG, PERg, S4, NG, PERg, Cdim, 1.0f, NF, NF, 0, NF, 0, 0, NF, 0, 0);
    gemm_wmma_kernel<EPI_NONE, 1><<<gp8, 256, 0, stream>>>(                     // phi_first
        cw_phi, Cdim, 0, 1, first, NG, PERg, S5, NG, PERg, Cdim, 1.0f, NF, NF, 0, NF, 0, 0, NF, 0, 0);
    gemm_wmma_kernel<EPI_NONE, 1><<<gp24, 256, 0, stream>>>(                    // phi_rest
        cw_phi, Cdim, 0, 1, rest, NG, PERg, S2, NG, PERg, Cdim, 1.0f, NF, NF, 0, NF, 0, 0, NF, 0, 0);

    row_reduce_kernel<<<8 * Cdim, 256, 0, stream>>>(k0mean, S3, NG, 1.0f / (float)NG);
    alpha_kernel<<<8, 1024, 0, stream>>>(alpha0, S3, k0mean, Cdim, NG);
    colscale_kernel<<<(8 * PERg) / 256, 256, 0, stream>>>(S3, S3, alpha0, Cdim, NG); // k0a in place
    row_reduce_kernel<<<8 * Cdim, 256, 0, stream>>>(ksum0, S3, NG, 1.0f);

    dim3 gkv8(Cdim / TN, Cdim / TM, 8);
    gemm_wmma_kernel<EPI_NONE, 0><<<gkv8, 256, 0, stream>>>(                    // kvT0
        S4, NG, PERg, BMOD_INF, S3, NG, PERg, SKV, Cdim, KVg, NG, rsC, NF, NF, 0, NF, 0, 0, NF, 0, 0);

    z_kernel<<<24, 1024, 0, stream>>>(z2, S1, ksum0, Cdim, NG, 8);

    gemm_wmma_kernel<EPI_ZPHI, 1><<<gp24, 256, 0, stream>>>(                    // y_cross -> S4
        SKV, Cdim, KVg, 8, S1, NG, PERg, S4, NG, PERg, Cdim, 1.0f,
        NF, z2, NG, S2, PERg, NG, NF, 0, 0);

    ymean_phifirst_kernel<<<(int)(PERg / 256), 256, 0, stream>>>(S3, S4, S5, 24, 8, PERg);

    // FFN on first 8 groups
    dim3 gtr3(NG / 32, Cdim / 32, 8);
    transpose_add_kernel<<<gtr3, 256, 0, stream>>>(S1, first, S3, Cdim, NG);    // hc_in (8,n,c)
    ln_rows_kernel<<<8 * NG, 256, 0, stream>>>(S3, S1, cg1, cb1, Cdim, 0);      // hcln
    dim3 gcf1(Fdim / TN, NG / TM, 8);
    gemm_wmma_kernel<EPI_BIAS_RELU, 0><<<gcf1, 256, 0, stream>>>(
        S3, Cdim, HLg, BMOD_INF, cf1w, Cdim, 0, S2, Fdim, H1g, Cdim, 1.0f,
        cf1b, NF, 0, NF, 0, 0, NF, 0, 0);
    dim3 gcf2(Cdim / TN, NG / TM, 8);
    gemm_wmma_kernel<EPI_BIAS_ADD_RES, 0><<<gcf2, 256, 0, stream>>>(
        S2, Fdim, H1g, BMOD_INF, cf2w, Fdim, 0, S1, Cdim, HLg, Fdim, 1.0f,
        cf2b, NF, 0, NF, 0, 0, S3, HLg, Cdim);
    ln_rows_kernel<<<8 * NG, 256, 0, stream>>>(S2, S1, cg2, cb2, Cdim, 1);      // hco (8,n,c)
    dim3 gtr4(Cdim / 32, NG / 32, 8);
    transpose_add_kernel<<<gtr4, 256, 0, stream>>>(S5, S2, nullptr, NG, Cdim);  // outfirst (8,c,n)

    final_kernel<<<((long long)Bsz * Cdim * Ndim) / 256, 256, 0, stream>>>(out, S5, S0);
}
